// DCGRUCell_22522808500420
// MI455X (gfx1250) — matrix operand order repaired and checked by reading, not compiled, not run on hardware
//
#include <hip/hip_runtime.h>
#include <hip/hip_bf16.h>

// DCGRU cell for MI455X (gfx1250).
// Heavy op: adj_t @ X  (8192x8192 @ 8192x72), 4x per call.
// Strategy: adjT in bf16 (fits 192MB L2), V_WMMA_F32_16X16X32_BF16 with f32
// accum, fold D^-1 and +I diagonal algebraically: adj_t @ x = A^T y + y,
// y = d_inv*x.  K split 4-way (2048 waves) so load latency is covered by
// co-resident waves; f32 partials combined in a cheap VALU pass.

#define NN     8192
#define BB     4
#define INDIM  2
#define UU     16
#define ITOT   18      // INDIM + UU
#define COLS   72      // ITOT * BB
#define CPAD   80      // padded to 5 WMMA col tiles
#define MDIFF  3       // K_DIFF + 1
#define KROWS  54      // ITOT * MDIFF
#define KSPLIT 4       // K-dimension split of the SpMM

typedef __bf16 bf16_t;
typedef __attribute__((ext_vector_type(16))) __bf16 v16bf;
typedef __attribute__((ext_vector_type(8)))  __bf16 v8bf;
typedef __attribute__((ext_vector_type(8)))  float  v8f;

union Frag16 { v16bf v; v8bf h[2]; };

// ---------------------------------------------------------------- row sums ->
// d_inv[j] = 1 / (1 + sum_i adj[j,i])   (the +1 is the folded identity diag)
__global__ void k_rowsum(const float* __restrict__ adj, float* __restrict__ dinv) {
    __shared__ float red[256];
    const int row = blockIdx.x;
    const float* p = adj + (size_t)row * NN;
    float s = 0.0f;
    for (int i = threadIdx.x; i < NN; i += 256) s += p[i];
    red[threadIdx.x] = s;
    __syncthreads();
    for (int w = 128; w > 0; w >>= 1) {
        if (threadIdx.x < w) red[threadIdx.x] += red[threadIdx.x + w];
        __syncthreads();
    }
    if (threadIdx.x == 0) {
        float d = red[0] + 1.0f;
        dinv[row] = (d > 0.0f) ? 1.0f / d : 0.0f;
    }
}

// ------------------------------------------------- adjT (bf16) via LDS tiles
__global__ void k_transpose_bf16(const float* __restrict__ adj, bf16_t* __restrict__ adjT) {
    __shared__ float tile[32][33];
    const int tx = threadIdx.x, ty = threadIdx.y;         // (32, 8)
    const int col0 = blockIdx.x * 32, row0 = blockIdx.y * 32;
    #pragma unroll
    for (int r = 0; r < 4; ++r)
        tile[ty + r * 8][tx] = adj[(size_t)(row0 + ty + r * 8) * NN + col0 + tx];
    __syncthreads();
    #pragma unroll
    for (int r = 0; r < 4; ++r)  // adjT[i*N + j] = adj[j*N + i]
        adjT[(size_t)(col0 + ty + r * 8) * NN + row0 + tx] = (bf16_t)tile[tx][ty + r * 8];
}

// ------------------------------------- build x0 [N,72] f32 and y0^T [80,N] bf16
// ru == nullptr: first gconv (concat(inputs, hx)); else second (concat(inputs, r*hx))
__global__ void k_build(const float* __restrict__ inputs, const float* __restrict__ hx,
                        const float* __restrict__ ru, const float* __restrict__ dinv,
                        float* __restrict__ x0, bf16_t* __restrict__ yT,
                        bf16_t* __restrict__ yT2) {
    const int n = blockIdx.x * blockDim.x + threadIdx.x;
    if (n >= NN) return;
    const float di = dinv[n];
    #pragma unroll
    for (int i = 0; i < ITOT; ++i) {
        #pragma unroll
        for (int b = 0; b < BB; ++b) {
            float v;
            if (i < INDIM) {
                v = inputs[(size_t)b * (NN * INDIM) + (size_t)n * INDIM + i];
            } else {
                float h = hx[(size_t)b * (NN * UU) + (size_t)n * UU + (i - INDIM)];
                if (ru) h *= ru[((size_t)b * NN + n) * 32 + (i - INDIM)];
                v = h;
            }
            const int c = i * BB + b;
            x0[(size_t)n * COLS + c] = v;
            yT[(size_t)c * NN + n] = (bf16_t)(di * v);
        }
    }
    #pragma unroll
    for (int c = COLS; c < CPAD; ++c) {            // zero pad columns of both y^T
        yT[(size_t)c * NN + n]  = (bf16_t)0.0f;
        yT2[(size_t)c * NN + n] = (bf16_t)0.0f;
    }
}

// -------------------------------------------------- SpMM partial (WMMA core)
// part[ks][i][c] = sum_{j in K-slice ks} adjT[i,j] * yT[c,j]
// One wave per block; wave owns a 16x80 D tile over a 2048-wide K slice.
__global__ void k_spmm_part(const bf16_t* __restrict__ adjT, const bf16_t* __restrict__ yT,
                            float* __restrict__ part) {
    const int i0   = blockIdx.x << 4;
    const int ks   = blockIdx.y;
    const int lane = threadIdx.x;          // blockDim.x == 32 -> EXEC all ones
    const int ll = lane & 15, lh = lane >> 4;
    const int kbeg = ks * (NN / KSPLIT);
    const int kend = kbeg + (NN / KSPLIT);

    // A fragment (16x32 bf16): lane ll row i0+ll; lanes 16-31 carry K offsets +8/+24
    const bf16_t* abase = adjT + (size_t)(i0 + ll) * NN + lh * 8;
    // B fragment (32x16 bf16): lane ll column; lanes 16-31 carry K offsets +8/+24
    const bf16_t* bbase = yT + (size_t)ll * NN + lh * 8;

    v8f acc[5] = {};
    for (int j0 = kbeg; j0 < kend; j0 += 32) {
        Frag16 a;
        a.h[0] = *(const v8bf*)(abase + j0);        // K = j0 + lh*8 + [0..7]
        a.h[1] = *(const v8bf*)(abase + j0 + 16);   // K = j0 + lh*8 + [16..23]
        #pragma unroll
        for (int t = 0; t < 5; ++t) {
            const bf16_t* bp = bbase + (size_t)t * 16 * NN + j0;
            Frag16 b;
            b.h[0] = *(const v8bf*)(bp);            // K = j0 + lh*8 + [0..7]
            b.h[1] = *(const v8bf*)(bp + 16);       // K = j0 + lh*8 + [16..23]
            acc[t] = __builtin_amdgcn_wmma_f32_16x16x32_bf16(
                false, a.v, false, b.v, (short)0, acc[t], false, false);
        }
    }

    // D layout: lane=(lh,ll), VGPR v -> element (M = v + 8*lh, Ncol = ll)
    float* pp = part + (size_t)ks * NN * CPAD;
    #pragma unroll
    for (int t = 0; t < 5; ++t) {
        #pragma unroll
        for (int v = 0; v < 8; ++v)
            pp[(size_t)(i0 + v + 8 * lh) * CPAD + t * 16 + ll] = acc[t][v];
    }
}

// --------------------------------------- combine partials + diffusion epilogue
// step 1:  xout = sum(part) + d_inv*xin,  emit y_next^T = bf16(d_inv * xout)
// step 2:  xout = 2*(sum(part) + d_inv*xin) - xprev
__global__ void k_combine(const float* __restrict__ part, const float* __restrict__ xin,
                          const float* __restrict__ xprev, const float* __restrict__ dinv,
                          float* __restrict__ xout, bf16_t* __restrict__ youtT, int step) {
    const int n = blockIdx.x * blockDim.x + threadIdx.x;
    if (n >= NN) return;
    const float di = dinv[n];
    #pragma unroll 8
    for (int c = 0; c < COLS; ++c) {
        float s = 0.0f;
        #pragma unroll
        for (int k = 0; k < KSPLIT; ++k)
            s += part[(size_t)k * NN * CPAD + (size_t)n * CPAD + c];
        float val = s + di * xin[(size_t)n * COLS + c];
        if (step == 2) val = 2.0f * val - xprev[(size_t)n * COLS + c];
        xout[(size_t)n * COLS + c] = val;
        if (step == 1) youtT[(size_t)c * NN + n] = (bf16_t)(di * val);
    }
}

// ------------------------------------------- ru = sigmoid(X @ W_ru + b_ru)
__global__ void k_gate_ru(const float* __restrict__ x0, const float* __restrict__ x1,
                          const float* __restrict__ x2, const float* __restrict__ W,
                          const float* __restrict__ bias, float* __restrict__ ru) {
    __shared__ float Ws[KROWS * 32];
    for (int t = threadIdx.x; t < KROWS * 32; t += blockDim.x) Ws[t] = W[t];
    __syncthreads();
    const int t = blockIdx.x * blockDim.x + threadIdx.x;   // t = b*N + n
    const int n = t & (NN - 1), b = t >> 13;
    const float* xs[3] = {x0, x1, x2};
    float xr[KROWS];
    #pragma unroll
    for (int m = 0; m < MDIFF; ++m)
        #pragma unroll
        for (int i = 0; i < ITOT; ++i)
            xr[i * MDIFF + m] = xs[m][(size_t)n * COLS + i * BB + b];
    #pragma unroll 4
    for (int o = 0; o < 32; ++o) {
        float s = bias[o];
        #pragma unroll
        for (int k = 0; k < KROWS; ++k) s += xr[k] * Ws[k * 32 + o];
        ru[(size_t)t * 32 + o] = 1.0f / (1.0f + expf(-s));
    }
}

// --------------- out = u*hx + (1-u)*tanh(X @ W_c + b_c)
__global__ void k_final(const float* __restrict__ x0, const float* __restrict__ x1,
                        const float* __restrict__ x2, const float* __restrict__ W,
                        const float* __restrict__ bias, const float* __restrict__ ru,
                        const float* __restrict__ hx, float* __restrict__ out) {
    __shared__ float Ws[KROWS * UU];
    for (int t = threadIdx.x; t < KROWS * UU; t += blockDim.x) Ws[t] = W[t];
    __syncthreads();
    const int t = blockIdx.x * blockDim.x + threadIdx.x;   // t = b*N + n
    const int n = t & (NN - 1), b = t >> 13;
    const float* xs[3] = {x0, x1, x2};
    float xr[KROWS];
    #pragma unroll
    for (int m = 0; m < MDIFF; ++m)
        #pragma unroll
        for (int i = 0; i < ITOT; ++i)
            xr[i * MDIFF + m] = xs[m][(size_t)n * COLS + i * BB + b];
    #pragma unroll 4
    for (int o = 0; o < UU; ++o) {
        float s = bias[o];
        #pragma unroll
        for (int k = 0; k < KROWS; ++k) s += xr[k] * Ws[k * UU + o];
        const float c = tanhf(s);
        const float u = ru[(size_t)t * 32 + 16 + o];
        const float h = hx[(size_t)b * (NN * UU) + (size_t)n * UU + o];
        out[(size_t)b * (NN * UU) + (size_t)n * UU + o] = u * h + (1.0f - u) * c;
    }
}

// =============================================================== launcher ====
extern "C" void kernel_launch(void* const* d_in, const int* in_sizes, int n_in,
                              void* d_out, int out_size, void* d_ws, size_t ws_size,
                              hipStream_t stream) {
    const float* inputs = (const float*)d_in[0];   // [B, N*INDIM]
    const float* hx     = (const float*)d_in[1];   // [B, N*U]
    const float* adj    = (const float*)d_in[2];   // [N, N]
    const float* W_ru   = (const float*)d_in[3];   // [54, 32]
    const float* b_ru   = (const float*)d_in[4];   // [32]
    const float* W_c    = (const float*)d_in[5];   // [54, 16]
    const float* b_c    = (const float*)d_in[6];   // [16]
    float* out = (float*)d_out;                    // [B, N*U]

    // workspace carve (all 256B aligned)
    char* ws = (char*)d_ws;
    size_t off = 0;
    bf16_t* adjT = (bf16_t*)(ws + off); off += (size_t)NN * NN * sizeof(bf16_t); // 128 MB
    float*  dinv = (float*)(ws + off);  off += (size_t)NN * sizeof(float);
    float*  xA   = (float*)(ws + off);  off += (size_t)NN * COLS * sizeof(float);
    float*  xB   = (float*)(ws + off);  off += (size_t)NN * COLS * sizeof(float);
    float*  xC   = (float*)(ws + off);  off += (size_t)NN * COLS * sizeof(float);
    bf16_t* y0T  = (bf16_t*)(ws + off); off += (size_t)CPAD * NN * sizeof(bf16_t);
    bf16_t* y1T  = (bf16_t*)(ws + off); off += (size_t)CPAD * NN * sizeof(bf16_t);
    float*  ru   = (float*)(ws + off);  off += (size_t)BB * NN * 32 * sizeof(float);
    float*  part = (float*)(ws + off);  off += (size_t)KSPLIT * NN * CPAD * sizeof(float);
    (void)ws_size; (void)in_sizes; (void)n_in; (void)out_size;

    const dim3 mmGrid(NN / 16, KSPLIT);

    // one-time per call: normalization + bf16 transpose of adjacency
    k_rowsum<<<NN, 256, 0, stream>>>(adj, dinv);
    k_transpose_bf16<<<dim3(NN / 32, NN / 32), dim3(32, 8), 0, stream>>>(adj, adjT);

    // ---- gconv #1 on concat(inputs, hx) ----
    k_build<<<NN / 256, 256, 0, stream>>>(inputs, hx, nullptr, dinv, xA, y0T, y1T);
    k_spmm_part<<<mmGrid, 32, 0, stream>>>(adjT, y0T, part);
    k_combine<<<NN / 256, 256, 0, stream>>>(part, xA, xA, dinv, xB, y1T, 1);
    k_spmm_part<<<mmGrid, 32, 0, stream>>>(adjT, y1T, part);
    k_combine<<<NN / 256, 256, 0, stream>>>(part, xB, xA, dinv, xC, y1T, 2);
    k_gate_ru<<<(BB * NN) / 256, 256, 0, stream>>>(xA, xB, xC, W_ru, b_ru, ru);

    // ---- gconv #2 on concat(inputs, r*hx) ----
    k_build<<<NN / 256, 256, 0, stream>>>(inputs, hx, ru, dinv, xA, y0T, y1T);
    k_spmm_part<<<mmGrid, 32, 0, stream>>>(adjT, y0T, part);
    k_combine<<<NN / 256, 256, 0, stream>>>(part, xA, xA, dinv, xB, y1T, 1);
    k_spmm_part<<<mmGrid, 32, 0, stream>>>(adjT, y1T, part);
    k_combine<<<NN / 256, 256, 0, stream>>>(part, xB, xA, dinv, xC, y1T, 2);
    k_final<<<(BB * NN) / 256, 256, 0, stream>>>(xA, xB, xC, W_c, b_c, ru, hx, out);
}
